// RAGAFAttentionModule_1486058684822
// MI455X (gfx1250) — matrix operand order repaired and checked by reading, not compiled, not run on hardware
//
#include <hip/hip_runtime.h>
#include <hip/hip_bf16.h>

// ---------------- problem constants ----------------
#define N_NODES 20000
#define N_EDGES 320000
#define T_TOK   77
#define HID     512
#define HEADS   8
#define HD      64
#define MPAD    20096   // 157 * 128
#define TPAD    128
#define LN_EPS  1e-5f
#define INV_SCALE 0.125f   // 1/sqrt(HD)

typedef __attribute__((ext_vector_type(16))) __bf16 v16bf;
typedef __attribute__((ext_vector_type(8)))  float  v8f;

// ---- gfx1250 async global->LDS copy (guarded; falls back to manual staging) ----
#if defined(__has_builtin)
#  if __has_builtin(__builtin_amdgcn_global_load_async_to_lds_b128)
#    define HAVE_ASYNC_COPY 1
#  endif
#  if __has_builtin(__builtin_amdgcn_s_wait_asynccnt)
#    define HAVE_WAIT_ASYNC 1
#  endif
#endif

// builtin signature (from hipcc diagnostic): param0 = v4i in addrspace(1) ("__device__"),
// non-const; use matching vector-pointer casts.
typedef int v4i_ __attribute__((__vector_size__(4 * sizeof(int))));
typedef __attribute__((address_space(1))) v4i_* gas_ptr;
typedef __attribute__((address_space(3))) v4i_* las_ptr;
#define ASG(p) ((gas_ptr)(p))
#define ASL(p) ((las_ptr)(p))

// ---------------- small device helpers ----------------
__device__ __forceinline__ unsigned short f2bf(float f) {
  unsigned u = __float_as_uint(f);
  u += 0x7FFFu + ((u >> 16) & 1u);     // round-to-nearest-even
  return (unsigned short)(u >> 16);
}
__device__ __forceinline__ unsigned ford(float f) {  // order-preserving f32->u32
  unsigned u = __float_as_uint(f);
  return (u & 0x80000000u) ? ~u : (u | 0x80000000u);
}
__device__ __forceinline__ float fdec(unsigned u) {
  unsigned v = (u & 0x80000000u) ? (u & 0x7FFFFFFFu) : ~u;
  return __uint_as_float(v);
}

// ---------------- utility kernels ----------------
__global__ void k_zero32(unsigned* __restrict__ p, long long n) {
  long long i = (long long)blockIdx.x * blockDim.x + threadIdx.x;
  if (i < n) p[i] = 0u;
}

__global__ void k_cvt(const float* __restrict__ in, unsigned short* __restrict__ out,
                      long long valid, long long total) {
  long long i = (long long)blockIdx.x * blockDim.x + threadIdx.x;
  if (i < total) out[i] = (i < valid) ? f2bf(in[i]) : (unsigned short)0;
}

// transpose-convert a weight matrix: in is K x Ncol (row-major f32),
// out is Ncol x K (row-major bf16) so GEMM B tiles are contiguous per output column.
__global__ void k_cvtw(const float* __restrict__ in, unsigned short* __restrict__ out,
                       int K, int Ncol) {
  long long i = (long long)blockIdx.x * blockDim.x + threadIdx.x;
  if (i >= (long long)K * Ncol) return;
  int k = (int)(i / Ncol), n = (int)(i % Ncol);
  out[(size_t)n * K + k] = f2bf(in[i]);
}

// x = node_features @ emb_w + emb_b    (K = 6, scalar path)
__global__ void k_embed(const float* __restrict__ nf, const float* __restrict__ w,
                        const float* __restrict__ b, float* __restrict__ x) {
  long long i = (long long)blockIdx.x * blockDim.x + threadIdx.x;
  if (i >= (long long)N_NODES * HID) return;
  int n = (int)(i / HID), h = (int)(i % HID);
  float acc = b[h];
#pragma unroll
  for (int d = 0; d < 6; ++d) acc += nf[n * 6 + d] * w[d * HID + h];
  x[i] = acc;
}

// ---------------- WMMA bf16 GEMM ----------------
// C(MxNcol) = A(MxK) @ B(KxNcol) + bias, with BT = B transposed (Ncol x K) bf16.
// Block: 256 threads = 8 waves; tile 128(M) x 128(N); wave -> 32(M) x 64(N) subtile
// (2 A-frags x 4 B-frags = 8 WMMA per 32-deep K step).
__global__ __launch_bounds__(256) void k_gemm_bf16(
    const unsigned short* __restrict__ A,    // M x K, bf16 (rows padded/allocated)
    const unsigned short* __restrict__ BT,   // Ncol x K, bf16 (pre-transposed weights)
    const float* __restrict__ bias,
    float* __restrict__ C,
    int M, int K, int Ncol)
{
  __shared__ unsigned short Als[128][40];   // stride 80B: 16B-aligned rows, conflict-free frags
  __shared__ unsigned short Bls[128][40];   // [n][k]

  const int tid   = threadIdx.x;
  const int wave  = tid >> 5;
  const int lane  = tid & 31;
  const int tileM = blockIdx.y * 128;
  const int tileN = blockIdx.x * 128;
  const int wm    = (wave >> 1) * 32;       // 0,32,64,96
  const int wn    = (wave & 1) * 64;        // 0,64

  v8f acc[2][4] = {};

  const int sr = tid >> 1;                  // staging: row 0..127 (A row / B out-col)
  const int sc = (tid & 1) * 16;            //          k-chunk 0/16 (halves)

  const int mloc = lane & 15;               // A frag: M = lane&15
  const int akb  = (lane >> 4) * 8;         //         k base 0/8, unused (A frag reads at bkb 0/16)
  const int bkb  = (lane >> 4) * 16;        // B frag: k base 0/16 per half-wave
  const int nloc = lane & 15;               //         N = lane&15

  const unsigned short* gA = A  + (size_t)(tileM + sr) * K + sc;
  const unsigned short* gB = BT + (size_t)(tileN + sr) * K + sc;

  for (int k0 = 0; k0 < K; k0 += 32) {
#if HAVE_ASYNC_COPY
    // ---- async global->LDS staging (ASYNCcnt path, no VGPR round-trip) ----
    __builtin_amdgcn_global_load_async_to_lds_b128(ASG(gA + k0), ASL(&Als[sr][sc]), 0,  0);
    __builtin_amdgcn_global_load_async_to_lds_b128(ASG(gA + k0), ASL(&Als[sr][sc]), 16, 0);
    __builtin_amdgcn_global_load_async_to_lds_b128(ASG(gB + k0), ASL(&Bls[sr][sc]), 0,  0);
    __builtin_amdgcn_global_load_async_to_lds_b128(ASG(gB + k0), ASL(&Bls[sr][sc]), 16, 0);
#  if HAVE_WAIT_ASYNC
    __builtin_amdgcn_s_wait_asynccnt(0);
#  else
    asm volatile("s_wait_asynccnt 0x0" ::: "memory");
#  endif
#else
    // ---- manual staging fallback ----
    {
      const uint4* sa = (const uint4*)(gA + k0);
      uint4 va0 = sa[0], va1 = sa[1];
      unsigned* da = (unsigned*)&Als[sr][sc];
      da[0] = va0.x; da[1] = va0.y; da[2] = va0.z; da[3] = va0.w;
      da[4] = va1.x; da[5] = va1.y; da[6] = va1.z; da[7] = va1.w;
      const uint4* sb = (const uint4*)(gB + k0);
      uint4 vb0 = sb[0], vb1 = sb[1];
      unsigned* db = (unsigned*)&Bls[sr][sc];
      db[0] = vb0.x; db[1] = vb0.y; db[2] = vb0.z; db[3] = vb0.w;
      db[4] = vb1.x; db[5] = vb1.y; db[6] = vb1.z; db[7] = vb1.w;
    }
#endif
    // prefetch next K slice (speculative; dropped past end)
    __builtin_prefetch(gA + k0 + 32, 0, 0);
    __builtin_prefetch(gB + k0 + 32, 0, 0);
    __syncthreads();

    // ---- fragments: A(16x32)/B(32x16) both in contiguous k = 16*(lane/16) + s order ----
    union Frag { v16bf v; unsigned u[8]; };
    Frag af[2], bf[4];
#pragma unroll
    for (int r = 0; r < 8; ++r) {
      int ka = bkb + 2 * r;
#pragma unroll
      for (int mi = 0; mi < 2; ++mi)
        af[mi].u[r] = *(const unsigned*)&Als[wm + mi * 16 + mloc][ka];
      int kb2 = bkb + 2 * r;
#pragma unroll
      for (int ni = 0; ni < 4; ++ni)
        bf[ni].u[r] = *(const unsigned*)&Bls[wn + ni * 16 + nloc][kb2];
    }

#pragma unroll
    for (int mi = 0; mi < 2; ++mi)
#pragma unroll
      for (int ni = 0; ni < 4; ++ni)
        acc[mi][ni] = __builtin_amdgcn_wmma_f32_16x16x32_bf16(
            false, af[mi].v, false, bf[ni].v, (short)0, acc[mi][ni], false, false);
    __syncthreads();
  }

  // ---- store (C/D layout: lanes 0-15 M=r, lanes 16-31 M=r+8; N=lane&15) ----
  const int mrow = (lane >> 4) * 8;
  const int ncol = lane & 15;
#pragma unroll
  for (int r = 0; r < 8; ++r) {
#pragma unroll
    for (int mi = 0; mi < 2; ++mi) {
      int row = tileM + wm + mi * 16 + mrow + r;
      if (row < M) {
#pragma unroll
        for (int ni = 0; ni < 4; ++ni) {
          int col = tileN + wn + ni * 16 + ncol;
          C[(size_t)row * Ncol + col] = acc[mi][ni][r] + bias[col];
        }
      }
    }
  }
}

// ---------------- graph attention: edge scores + segment max ----------------
__global__ void k_scores(const float* __restrict__ Q, const float* __restrict__ Kf,
                         const int* __restrict__ src, const int* __restrict__ tgt,
                         const float* __restrict__ ew, const float* __restrict__ e_w,
                         const float* __restrict__ e_b,
                         float* __restrict__ sc, unsigned* __restrict__ nmax)
{
  int i = blockIdx.x * blockDim.x + threadIdx.x;
  if (i >= N_EDGES * HEADS) return;
  int e = i >> 3, h = i & 7;
  int s = src[e], t = tgt[e];
  const float* qr = Q  + (size_t)t * HID + h * HD;
  const float* kr = Kf + (size_t)s * HID + h * HD;
  float d = 0.f;
#pragma unroll 8
  for (int j = 0; j < HD; ++j) d += qr[j] * kr[j];
  float v = d * INV_SCALE + ew[e] * e_w[h] + e_b[h];
  sc[i] = v;
  atomicMax(&nmax[(size_t)t * HEADS + h], ford(v));
}

__global__ void k_expsum(float* __restrict__ sc, const int* __restrict__ tgt,
                         const unsigned* __restrict__ nmax, float* __restrict__ nsum)
{
  int i = blockIdx.x * blockDim.x + threadIdx.x;
  if (i >= N_EDGES * HEADS) return;
  int e = i >> 3, h = i & 7;
  int t = tgt[e];
  float m  = fdec(nmax[(size_t)t * HEADS + h]);
  float ex = __expf(sc[i] - m);
  sc[i] = ex;
  atomicAdd(&nsum[(size_t)t * HEADS + h], ex);
}

__global__ void k_agg(const float* __restrict__ sc, const int* __restrict__ src,
                      const int* __restrict__ tgt, const float* __restrict__ nsum,
                      const float* __restrict__ V, float* __restrict__ agg)
{
  int i = blockIdx.x * blockDim.x + threadIdx.x;
  if (i >= N_EDGES * HEADS) return;
  int e = i >> 3, h = i & 7;
  int s = src[e], t = tgt[e];
  float p = sc[i] / (nsum[(size_t)t * HEADS + h] + 1e-8f);
  const float* vr = V   + (size_t)s * HID + h * HD;
  float*       ar = agg + (size_t)t * HID + h * HD;
  for (int j = 0; j < HD; ++j) atomicAdd(&ar[j], p * vr[j]);
}

// ---------------- residual + layernorm (block per node) ----------------
__global__ __launch_bounds__(256) void k_addln(float* __restrict__ x, const float* __restrict__ g,
                                               const float* __restrict__ gam, const float* __restrict__ bet)
{
  __shared__ float red[256];
  int n = blockIdx.x, tid = threadIdx.x;
  size_t base = (size_t)n * HID;
  float a0 = x[base + tid]       + g[base + tid];
  float a1 = x[base + tid + 256] + g[base + tid + 256];
  red[tid] = a0 + a1;
  __syncthreads();
  for (int o = 128; o > 0; o >>= 1) { if (tid < o) red[tid] += red[tid + o]; __syncthreads(); }
  float mu = red[0] * (1.0f / HID);
  __syncthreads();
  float d0 = a0 - mu, d1 = a1 - mu;
  red[tid] = d0 * d0 + d1 * d1;
  __syncthreads();
  for (int o = 128; o > 0; o >>= 1) { if (tid < o) red[tid] += red[tid + o]; __syncthreads(); }
  float rs = rsqrtf(red[0] * (1.0f / HID) + LN_EPS);
  x[base + tid]       = d0 * rs * gam[tid]       + bet[tid];
  x[base + tid + 256] = d1 * rs * gam[tid + 256] + bet[tid + 256];
}

// ---------------- cross attention: wave per (node, head) ----------------
__global__ __launch_bounds__(256) void k_cross(const float* __restrict__ Q,
                                               const float* __restrict__ KC,
                                               const float* __restrict__ VC,
                                               float* __restrict__ OutC,
                                               float* __restrict__ attn)
{
  int n = blockIdx.x;
  int h = threadIdx.x >> 5;     // 8 waves == 8 heads
  int lane = threadIdx.x & 31;
  const float* q = Q + (size_t)n * HID + h * HD;
  float q0 = q[lane], q1 = q[lane + 32];

  float mx = -1e30f;
  for (int t = 0; t < T_TOK; ++t) {
    const float* kr = KC + (size_t)t * HID + h * HD;
    float p = q0 * kr[lane] + q1 * kr[lane + 32];
    for (int o = 16; o > 0; o >>= 1) p += __shfl_xor(p, o, 32);
    mx = fmaxf(mx, p * INV_SCALE);
  }
  float sum = 0.f;
  for (int t = 0; t < T_TOK; ++t) {
    const float* kr = KC + (size_t)t * HID + h * HD;
    float p = q0 * kr[lane] + q1 * kr[lane + 32];
    for (int o = 16; o > 0; o >>= 1) p += __shfl_xor(p, o, 32);
    sum += __expf(p * INV_SCALE - mx);
  }
  float inv = 1.0f / sum;
  float o0 = 0.f, o1 = 0.f;
  for (int t = 0; t < T_TOK; ++t) {
    const float* kr = KC + (size_t)t * HID + h * HD;
    float p = q0 * kr[lane] + q1 * kr[lane + 32];
    for (int o = 16; o > 0; o >>= 1) p += __shfl_xor(p, o, 32);
    float pr = __expf(p * INV_SCALE - mx) * inv;
    const float* vr = VC + (size_t)t * HID + h * HD;
    o0 += pr * vr[lane];
    o1 += pr * vr[lane + 32];
    if (lane == 0) atomicAdd(&attn[(size_t)n * T_TOK + t], pr * (1.0f / HEADS));
  }
  OutC[(size_t)n * HID + h * HD + lane]      = o0;
  OutC[(size_t)n * HID + h * HD + lane + 32] = o1;
}

// ---------------- host orchestration ----------------
extern "C" void kernel_launch(void* const* d_in, const int* in_sizes, int n_in,
                              void* d_out, int out_size, void* d_ws, size_t ws_size,
                              hipStream_t stream) {
  (void)in_sizes; (void)n_in; (void)out_size; (void)ws_size;

  const float* nf  = (const float*)d_in[0];
  const int*   ei  = (const int*)d_in[1];
  const float* ew  = (const float*)d_in[2];
  const float* txt = (const float*)d_in[3];
  const int* src = ei;
  const int* tgt = ei + N_EDGES;

  // jax pytree (sorted dict keys) leaf order
  int pi = 4;
  auto P = [&](void) { return (const float*)d_in[pi++]; };
  const float* c_k_b = P();      const float* c_k_w = P();
  const float* c_o_b = P();      const float* c_o_w = P();
  const float* c_q_b = P();      const float* c_q_w = P();
  const float* c_region_b = P(); const float* c_region_w = P();
  const float* c_text_b = P();   const float* c_text_w = P();
  const float* c_v_b = P();      const float* c_v_w = P();
  const float* cross_ln_b = P(); const float* cross_ln_g = P();
  const float* emb_b = P();      const float* emb_w = P();
  struct Lp { const float *e_b,*e_w,*k_b,*k_w,*ln_b,*ln_g,*o_b,*o_w,*q_b,*q_w,*v_b,*v_w; } L[2];
  for (int l = 0; l < 2; ++l) {
    L[l].e_b = P(); L[l].e_w = P(); L[l].k_b = P(); L[l].k_w = P();
    L[l].ln_b = P(); L[l].ln_g = P(); L[l].o_b = P(); L[l].o_w = P();
    L[l].q_b = P(); L[l].q_w = P(); L[l].v_b = P(); L[l].v_w = P();
  }
  const float* out_b = P(); const float* out_w = P();

  // ---- carve workspace (256B aligned) ----
  char* w = (char*)d_ws;
  size_t off = 0;
  auto carve = [&](size_t bytes) { void* p = w + off; off += (bytes + 255) & ~(size_t)255; return p; };
  float*          X    = (float*)carve((size_t)MPAD * HID * 4);
  float*          Qb   = (float*)carve((size_t)MPAD * HID * 4);
  float*          Kb   = (float*)carve((size_t)MPAD * HID * 4);
  float*          Vb   = (float*)carve((size_t)MPAD * HID * 4);
  float*          Db   = (float*)carve((size_t)MPAD * HID * 4);
  float*          SC   = (float*)carve((size_t)N_EDGES * HEADS * 4);
  unsigned*       NM   = (unsigned*)carve((size_t)N_NODES * HEADS * 4);
  float*          NS   = (float*)carve((size_t)N_NODES * HEADS * 4);
  unsigned short* XBF  = (unsigned short*)carve((size_t)MPAD * HID * 2);
  unsigned short* WBF  = (unsigned short*)carve((size_t)768 * HID * 2);   // transposed weights
  float*          TP   = (float*)carve((size_t)TPAD * HID * 4);
  float*          KC   = (float*)carve((size_t)TPAD * HID * 4);
  float*          VC   = (float*)carve((size_t)TPAD * HID * 4);
  unsigned short* TPBF = (unsigned short*)carve((size_t)TPAD * HID * 2);
  unsigned short* TXBF = (unsigned short*)carve((size_t)TPAD * 768 * 2);

  const long long NH = (long long)N_NODES * HID;
  const long long MH = (long long)MPAD * HID;
  const long long EH = (long long)N_EDGES * HEADS;
  const int EB = (int)((EH + 255) / 256);

  auto cvt = [&](const float* s, unsigned short* d, long long valid, long long total) {
    k_cvt<<<(int)((total + 255) / 256), 256, 0, stream>>>(s, d, valid, total);
  };
  auto cvtw = [&](const float* s, int K) {   // weights K x 512 -> WBF 512 x K (bf16)
    long long n = (long long)K * HID;
    k_cvtw<<<(int)((n + 255) / 256), 256, 0, stream>>>(s, WBF, K, HID);
  };
  auto zero32 = [&](void* p, long long n) {
    k_zero32<<<(int)((n + 255) / 256), 256, 0, stream>>>((unsigned*)p, n);
  };
  auto gemm = [&](const unsigned short* Ab, const float* bias, float* Cc,
                  int M, int Mp, int Kd) {
    dim3 g(HID / 128, Mp / 128);
    k_gemm_bf16<<<g, 256, 0, stream>>>(Ab, WBF, bias, Cc, M, Kd, HID);
  };

  // ---- embed ----
  k_embed<<<(int)((NH + 255) / 256), 256, 0, stream>>>(nf, emb_w, emb_b, X);

  // ---- graph attention layers ----
  for (int l = 0; l < 2; ++l) {
    cvt(X, XBF, NH, MH);
    cvtw(L[l].q_w, HID); gemm(XBF, L[l].q_b, Qb, N_NODES, MPAD, HID);
    cvtw(L[l].k_w, HID); gemm(XBF, L[l].k_b, Kb, N_NODES, MPAD, HID);
    cvtw(L[l].v_w, HID); gemm(XBF, L[l].v_b, Vb, N_NODES, MPAD, HID);

    zero32(NM, (long long)N_NODES * HEADS);
    zero32(NS, (long long)N_NODES * HEADS);
    k_scores<<<EB, 256, 0, stream>>>(Qb, Kb, src, tgt, ew, L[l].e_w, L[l].e_b, SC, NM);
    k_expsum<<<EB, 256, 0, stream>>>(SC, tgt, NM, NS);
    zero32(Db, MH);
    k_agg<<<EB, 256, 0, stream>>>(SC, src, tgt, NS, Vb, Db);

    cvt(Db, XBF, MH, MH);
    cvtw(L[l].o_w, HID); gemm(XBF, L[l].o_b, Qb, N_NODES, MPAD, HID);
    k_addln<<<N_NODES, 256, 0, stream>>>(X, Qb, L[l].ln_g, L[l].ln_b);
  }

  // ---- cross attention ----
  cvt(X, XBF, NH, MH);
  cvtw(c_region_w, HID);
  gemm(XBF, c_region_b, Db, N_NODES, MPAD, HID);                      // r

  cvt(txt, TXBF, (long long)T_TOK * 768, (long long)TPAD * 768);
  cvtw(c_text_w, 768);
  gemm(TXBF, c_text_b, TP, T_TOK, TPAD, 768);                         // tp

  cvt(Db, XBF, NH, MH);
  cvtw(c_q_w, HID);
  gemm(XBF, c_q_b, Qb, N_NODES, MPAD, HID);                           // Qc

  cvt(TP, TPBF, (long long)T_TOK * HID, (long long)TPAD * HID);
  cvtw(c_k_w, HID); gemm(TPBF, c_k_b, KC, T_TOK, TPAD, HID);
  cvtw(c_v_w, HID); gemm(TPBF, c_v_b, VC, T_TOK, TPAD, HID);

  float* attn = (float*)d_out + (size_t)N_NODES * HID;
  zero32(attn, (long long)N_NODES * T_TOK);
  k_cross<<<N_NODES, 256, 0, stream>>>(Qb, KC, VC, Db, attn);

  cvt(Db, XBF, NH, MH);
  cvtw(c_o_w, HID);
  gemm(XBF, c_o_b, Qb, N_NODES, MPAD, HID);                           // co
  k_addln<<<N_NODES, 256, 0, stream>>>(X, Qb, cross_ln_g, cross_ln_b);

  cvt(X, XBF, NH, MH);
  cvtw(out_w, HID);
  gemm(XBF, out_b, (float*)d_out, N_NODES, MPAD, HID);                // final out
}
